// EnergyPredTransformerGNN_6270652252425
// MI455X (gfx1250) — matrix operand order repaired and checked by reading, not compiled, not run on hardware
//
#include <hip/hip_runtime.h>
#include <hip/hip_bf16.h>
#include <math.h>

// ---------------------------------------------------------------------------
// EnergyPredTransformerGNN forward for MI455X (gfx1250, wave32, WMMA).
//
// Pipeline per layer (single fused activation buffer qkvs[N,256] = Q|K|V|S):
//   1) init (amax=-inf, den=0, BN sums=0)
//   2) gemm_qkvs : qkvs = X @ [Wq|Wk|Wv|Ws] + bias  via v_wmma_f32_16x16x32_f16
//                  (S slice doubles as the skip-initialized aggregation buffer)
//   3) edge_logits : alpha = (q[dst]·(k[src]+ew*We))/4 ; segment-max (punned atomics)
//   4) edge_exp    : ex = exp(alpha - amax[dst]) ; segment-sum den
//   5) edge_scatter: S[dst] += (v[src]+ew*We) * ex/den   (atomic f32 adds)
//   6) bn_stats / bn_apply : x' = relu(BN(S))
// Then segment-mean pool over graphs and a tiny 2-layer FC head.
//
// d_in flatten-order assumption (sorted-pytree params after positional tensors):
//   [0] x  [1] edge_index  [2] edge_weight  [3] batch
//   [4] fc_W1 [5] fc_b1 [6] fc_W2 [7] fc_b2
//   [8+11*l + k], k in: We, Wk, Wq, Ws, Wv, bk, bn_b, bn_g, bq, bs, bv
// ---------------------------------------------------------------------------

typedef __attribute__((ext_vector_type(16))) _Float16 v16h;
typedef __attribute__((ext_vector_type(8)))  float    v8f;
typedef __attribute__((ext_vector_type(4)))  _Float16 h4;

#define HCDIM 64
#define NHEAD 4
#define CPH   16
#define GRAPHS 32
#define EPS_BN 1e-5f
#define QOFF 0
#define KOFF 64
#define VOFF 128
#define SOFF 192

__device__ __forceinline__ void atomicMaxF32(float* addr, float v) {
    // standard bit-punned float max: works for mixed-sign values with -inf init
    if (v >= 0.0f) atomicMax(reinterpret_cast<int*>(addr), __float_as_int(v));
    else           atomicMin(reinterpret_cast<unsigned int*>(addr), __float_as_uint(v));
}

// ---------------------------------------------------------------------------
// 1) per-layer init: amax=-inf, den=0, BN col sums=0
// ---------------------------------------------------------------------------
__global__ __launch_bounds__(256) void init_layer_kernel(float* __restrict__ amax,
                                                         float* __restrict__ den,
                                                         float* __restrict__ sums,
                                                         int NH) {
    int i = blockIdx.x * 256 + threadIdx.x;
    if (i < NH) {
        amax[i] = __int_as_float(0xFF800000);  // -inf
        den[i]  = 0.0f;
    }
    if (i < 2 * HCDIM) sums[i] = 0.0f;
}

// ---------------------------------------------------------------------------
// 2) fused QKVS GEMM:  qkvs[N,256] = X[N,64] @ [Wq|Wk|Wv|Ws] + [bq|bk|bv|bs]
//    f16 inputs, f32 accumulate, v_wmma_f32_16x16x32_f16.
//    Block: 256 thr = 8 waves; wave w owns col tiles {2w,2w+1}; loops row tiles.
//    Compile-time output stride (256) -> immediate-offset stores from one base.
// ---------------------------------------------------------------------------
__global__ __launch_bounds__(256)
void gemm_qkvs_kernel(const float* __restrict__ X,
                      const float* __restrict__ Wq, const float* __restrict__ Wk,
                      const float* __restrict__ Wv, const float* __restrict__ Ws,
                      const float* __restrict__ bq, const float* __restrict__ bk,
                      const float* __restrict__ bv, const float* __restrict__ bs,
                      float* __restrict__ qkvs,  // [N,256] = Q|K|V|S
                      int nTiles, int tilesPerBlock) {
    __shared__ _Float16 Wlds[64][256];     // concatenated weight panel, f16
    __shared__ _Float16 Xlds[16][72];      // padded 16x64 activation tile

    const int t    = threadIdx.x;
    const int wave = t >> 5;
    const int lane = t & 31;

    // stage weights (f32 -> f16), vectorized: 4-aligned runs never cross a
    // 64-col source-matrix boundary.
    for (int i = t * 4; i < 64 * 256; i += 256 * 4) {
        int k = i >> 8, n = i & 255;
        const float* Wp = (n < 64) ? Wq : (n < 128) ? Wk : (n < 192) ? Wv : Ws;
        float4 w4 = *reinterpret_cast<const float4*>(Wp + k * 64 + (n & 63));
        h4 hv;
        hv[0] = (_Float16)w4.x; hv[1] = (_Float16)w4.y;
        hv[2] = (_Float16)w4.z; hv[3] = (_Float16)w4.w;
        *reinterpret_cast<h4*>(&Wlds[k][n]) = hv;
    }
    __syncthreads();

    // Build B fragments once (live across the row loop).
    // B 32x16 f16 layout: lane&15 = N column; elements 0..7 = K khalf*8+0..7,
    // elements 8..15 = K 16+khalf*8+0..7.
    const int ncol  = lane & 15;
    const int khalf = lane >> 4;
    const int nb0   = (2 * wave) * 16;
    const int nb1   = nb0 + 16;
    v16h bf00, bf01, bf10, bf11;
    #pragma unroll
    for (int j = 0; j < 16; ++j) {
        int k0 = (j < 8) ? (khalf * 8 + j) : (16 + khalf * 8 + (j - 8));       // K-step 0 (K 0..31)
        int k1 = 32 + ((j < 8) ? (khalf * 8 + j) : (16 + khalf * 8 + (j - 8)));  // K-step 1 (K 32..63)
        bf00[j] = Wlds[k0][nb0 + ncol];
        bf01[j] = Wlds[k0][nb1 + ncol];
        bf10[j] = Wlds[k1][nb0 + ncol];
        bf11[j] = Wlds[k1][nb1 + ncol];
    }

    // per-lane biases for the two owned columns
    const float* bp0 = (nb0 < 64) ? bq : (nb0 < 128) ? bk : (nb0 < 192) ? bv : bs;
    const float* bp1 = (nb1 < 64) ? bq : (nb1 < 128) ? bk : (nb1 < 192) ? bv : bs;
    const float bias0 = bp0[(nb0 & 63) + ncol];
    const float bias1 = bp1[(nb1 & 63) + ncol];

    // per-thread X-staging coordinates: thread covers one float4 of the tile
    const int xr = t >> 4;
    const int xc = (t & 15) * 4;

    const int tile0 = blockIdx.x * tilesPerBlock;
    for (int rt = 0; rt < tilesPerBlock; ++rt) {
        int tile = tile0 + rt;
        if (tile >= nTiles) break;
        int rowBase = tile * 16;

        __syncthreads();  // protect Xlds from previous iteration consumers
        {
            float4 xv = *reinterpret_cast<const float4*>(
                X + (size_t)(rowBase + xr) * 64 + xc);
            h4 hv;
            hv[0] = (_Float16)xv.x; hv[1] = (_Float16)xv.y;
            hv[2] = (_Float16)xv.z; hv[3] = (_Float16)xv.w;
            *reinterpret_cast<h4*>(&Xlds[xr][xc]) = hv;
        }
        // prefetch next row tile while this one is consumed
        if (rt + 1 < tilesPerBlock && tile + 1 < nTiles) {
            __builtin_prefetch(X + (size_t)(rowBase + 16 + xr) * 64 + xc, 0, 1);
        }
        __syncthreads();

        // A 16x32 f16 layout: lane&15 = M row; elements 0..7 = K khalf*8+0..7,
        // elements 8..15 = K 16+khalf*8+0..7.
        const int m = lane & 15;
        v16h a0, a1;
        #pragma unroll
        for (int j = 0; j < 8; ++j) {
            a0[j]     = Xlds[m][khalf * 8 + j];
            a0[8 + j] = Xlds[m][16 + khalf * 8 + j];
            a1[j]     = Xlds[m][32 + khalf * 8 + j];
            a1[8 + j] = Xlds[m][48 + khalf * 8 + j];
        }

        v8f accA = {}; v8f accB = {};
        accA = __builtin_amdgcn_wmma_f32_16x16x32_f16(false, a0, false, bf00, (short)0, accA, false, false);
        accA = __builtin_amdgcn_wmma_f32_16x16x32_f16(false, a1, false, bf10, (short)0, accA, false, false);
        accB = __builtin_amdgcn_wmma_f32_16x16x32_f16(false, a0, false, bf01, (short)0, accB, false, false);
        accB = __builtin_amdgcn_wmma_f32_16x16x32_f16(false, a1, false, bf11, (short)0, accB, false, false);

        // D layout: element r -> M = r + 8*khalf, N = lane&15.
        // Single base pointer, stride 256 floats; both col tiles 16 apart.
        float* p = qkvs + ((size_t)(rowBase + 8 * khalf) << 8) + nb0 + ncol;
        #pragma unroll
        for (int r = 0; r < 8; ++r) {
            p[0]  = accA[r] + bias0;
            p[16] = accB[r] + bias1;
            p += 256;
        }
    }
}

// ---------------------------------------------------------------------------
// 3) attention logits + segment max.   one thread per (edge, head)
// ---------------------------------------------------------------------------
__global__ __launch_bounds__(256)
void edge_logits_kernel(const int* __restrict__ ei, const float* __restrict__ ew,
                        const float* __restrict__ qkvs, const float* __restrict__ We,
                        float* __restrict__ alpha, float* __restrict__ amax,
                        int E_) {
    int idx = blockIdx.x * 256 + threadIdx.x;
    if (idx >= E_ * NHEAD) return;
    int e = idx >> 2, h = idx & 3;
    int src = ei[e], dst = ei[E_ + e];
    float w = ew[e];
    const float4* q4 = reinterpret_cast<const float4*>(qkvs + ((size_t)dst << 8) + QOFF + h * CPH);
    const float4* k4 = reinterpret_cast<const float4*>(qkvs + ((size_t)src << 8) + KOFF + h * CPH);
    const float4* e4 = reinterpret_cast<const float4*>(We + h * CPH);
    float acc = 0.0f;
    #pragma unroll
    for (int i = 0; i < 4; ++i) {
        float4 q = q4[i], k = k4[i], ee = e4[i];
        acc += q.x * fmaf(w, ee.x, k.x);
        acc += q.y * fmaf(w, ee.y, k.y);
        acc += q.z * fmaf(w, ee.z, k.z);
        acc += q.w * fmaf(w, ee.w, k.w);
    }
    acc *= 0.25f;  // 1/sqrt(C), C=16
    alpha[idx] = acc;
    atomicMaxF32(&amax[dst * NHEAD + h], acc);
}

// ---------------------------------------------------------------------------
// 4) exp(alpha - amax[dst]) + segment sum
// ---------------------------------------------------------------------------
__global__ __launch_bounds__(256)
void edge_exp_kernel(const int* __restrict__ ei, float* __restrict__ alpha,
                     const float* __restrict__ amax, float* __restrict__ den,
                     int E_) {
    int idx = blockIdx.x * 256 + threadIdx.x;
    if (idx >= E_ * NHEAD) return;
    int e = idx >> 2, h = idx & 3;
    int dst = ei[E_ + e];
    float ex = __expf(alpha[idx] - amax[dst * NHEAD + h]);
    alpha[idx] = ex;  // store numerator in place
    atomicAdd(&den[dst * NHEAD + h], ex);
}

// ---------------------------------------------------------------------------
// 5) weighted message scatter:  S[dst] += (v[src]+ew*We) * ex/den[dst]
// ---------------------------------------------------------------------------
__global__ __launch_bounds__(256)
void edge_scatter_kernel(const int* __restrict__ ei, const float* __restrict__ ew,
                         float* __restrict__ qkvs, const float* __restrict__ We,
                         const float* __restrict__ alpha, const float* __restrict__ den,
                         int E_) {
    int idx = blockIdx.x * 256 + threadIdx.x;
    if (idx >= E_ * NHEAD) return;
    int e = idx >> 2, h = idx & 3;
    int src = ei[e], dst = ei[E_ + e];
    float w = ew[e];
    float a = alpha[idx] / fmaxf(den[dst * NHEAD + h], 1e-16f);
    const float4* v4 = reinterpret_cast<const float4*>(qkvs + ((size_t)src << 8) + VOFF + h * CPH);
    const float4* e4 = reinterpret_cast<const float4*>(We + h * CPH);
    float* o = qkvs + ((size_t)dst << 8) + SOFF + h * CPH;
    #pragma unroll
    for (int i = 0; i < 4; ++i) {
        float4 v = v4[i], ee = e4[i];
        atomicAdd(&o[i * 4 + 0], fmaf(w, ee.x, v.x) * a);
        atomicAdd(&o[i * 4 + 1], fmaf(w, ee.y, v.y) * a);
        atomicAdd(&o[i * 4 + 2], fmaf(w, ee.z, v.z) * a);
        atomicAdd(&o[i * 4 + 3], fmaf(w, ee.w, v.w) * a);
    }
}

// ---------------------------------------------------------------------------
// 6a) BN column sums (sum / sumsq) over N rows of the S slice
// ---------------------------------------------------------------------------
__global__ __launch_bounds__(256)
void bn_stats_kernel(const float* __restrict__ qkvs, float* __restrict__ sums,
                     int N_) {
    __shared__ float s1[256], s2[256];
    int c = threadIdx.x & 63;
    int rg = threadIdx.x >> 6;  // 0..3
    int row0 = blockIdx.x * 1024;
    int rowEnd = row0 + 1024; if (rowEnd > N_) rowEnd = N_;
    float a = 0.0f, b = 0.0f;
    for (int r = row0 + rg; r < rowEnd; r += 4) {
        float x = qkvs[((size_t)r << 8) + SOFF + c];
        a += x; b += x * x;
    }
    s1[threadIdx.x] = a; s2[threadIdx.x] = b;
    __syncthreads();
    if (rg == 0) {
        a = s1[c] + s1[64 + c] + s1[128 + c] + s1[192 + c];
        b = s2[c] + s2[64 + c] + s2[128 + c] + s2[192 + c];
        atomicAdd(&sums[c], a);
        atomicAdd(&sums[64 + c], b);
    }
}

// ---------------------------------------------------------------------------
// 6b) BN apply + ReLU -> next-layer activations [N,64]
// ---------------------------------------------------------------------------
__global__ __launch_bounds__(256)
void bn_apply_kernel(const float* __restrict__ qkvs, const float* __restrict__ sums,
                     const float* __restrict__ g, const float* __restrict__ b,
                     float* __restrict__ Xout, int N_) {
    int idx = blockIdx.x * 256 + threadIdx.x;
    if (idx >= N_ * 64) return;
    int n = idx >> 6, c = idx & 63;
    float invN = 1.0f / (float)N_;
    float mean = sums[c] * invN;
    float var  = sums[64 + c] * invN - mean * mean;
    float x = qkvs[((size_t)n << 8) + SOFF + c];
    float y = g[c] * (x - mean) * rsqrtf(var + EPS_BN) + b[c];
    Xout[idx] = fmaxf(y, 0.0f);
}

// ---------------------------------------------------------------------------
// pooling
// ---------------------------------------------------------------------------
__global__ __launch_bounds__(256)
void init_pool_kernel(float* __restrict__ pooled, float* __restrict__ cnt) {
    int i = blockIdx.x * 256 + threadIdx.x;
    if (i < GRAPHS * 64) pooled[i] = 0.0f;
    if (i < GRAPHS) cnt[i] = 0.0f;
}

__global__ __launch_bounds__(256)
void pool_kernel(const float* __restrict__ X, const int* __restrict__ batch,
                 float* __restrict__ pooled, float* __restrict__ cnt, int N_) {
    int idx = blockIdx.x * 256 + threadIdx.x;
    if (idx >= N_ * 64) return;
    int n = idx >> 6, c = idx & 63;
    int g = batch[n];
    atomicAdd(&pooled[g * 64 + c], X[idx]);
    if (c == 0) atomicAdd(&cnt[g], 1.0f);
}

// ---------------------------------------------------------------------------
// final FC head (32x64 -> 32x64 -> 32), single block
// ---------------------------------------------------------------------------
__global__ __launch_bounds__(256)
void head_kernel(const float* __restrict__ pooled, const float* __restrict__ cnt,
                 const float* __restrict__ W1, const float* __restrict__ b1,
                 const float* __restrict__ W2, const float* __restrict__ b2,
                 float* __restrict__ out) {
    __shared__ float pm[GRAPHS * 64];
    __shared__ float hb[GRAPHS * 64];
    int t = threadIdx.x;
    for (int i = t; i < GRAPHS * 64; i += 256) {
        int g = i >> 6;
        pm[i] = pooled[i] / fmaxf(cnt[g], 1.0f);
    }
    __syncthreads();
    for (int i = t; i < GRAPHS * 64; i += 256) {
        int g = i >> 6, j = i & 63;
        float acc = b1[j];
        for (int c = 0; c < 64; ++c) acc = fmaf(pm[g * 64 + c], W1[c * 64 + j], acc);
        hb[i] = fmaxf(acc, 0.0f);
    }
    __syncthreads();
    if (t < GRAPHS) {
        float acc = b2[0];
        for (int j = 0; j < 64; ++j) acc = fmaf(hb[t * 64 + j], W2[j], acc);
        out[t] = acc;
    }
}

// ---------------------------------------------------------------------------
// host launcher
// ---------------------------------------------------------------------------
extern "C" void kernel_launch(void* const* d_in, const int* in_sizes, int n_in,
                              void* d_out, int out_size, void* d_ws, size_t ws_size,
                              hipStream_t stream) {
    const float* x     = (const float*)d_in[0];
    const int*   ei    = (const int*)  d_in[1];
    const float* ew    = (const float*)d_in[2];
    const int*   batch = (const int*)  d_in[3];
    const float* fcW1  = (const float*)d_in[4];
    const float* fcb1  = (const float*)d_in[5];
    const float* fcW2  = (const float*)d_in[6];
    const float* fcb2  = (const float*)d_in[7];

    const int E_ = in_sizes[2];   // edge_weight count
    const int N_ = in_sizes[3];   // batch count

    // workspace carve-out (256B aligned)
    char* ws = (char*)d_ws;
    size_t off = 0;
    auto carve = [&](size_t bytes) -> float* {
        off = (off + 255) & ~(size_t)255;
        float* p = (float*)(ws + off);
        off += bytes;
        return p;
    };
    float* Xa     = carve((size_t)N_ * 64 * 4);
    float* Xb     = carve((size_t)N_ * 64 * 4);
    float* qkvs   = carve((size_t)N_ * 256 * 4);   // Q|K|V|S fused
    float* alpha  = carve((size_t)E_ * NHEAD * 4);
    float* amax   = carve((size_t)N_ * NHEAD * 4);
    float* den    = carve((size_t)N_ * NHEAD * 4);
    float* sums   = carve(2 * 64 * 4);
    float* pooled = carve(GRAPHS * 64 * 4);
    float* cnt    = carve(GRAPHS * 4);
    (void)ws_size; (void)n_in;

    const int NH  = N_ * NHEAD;
    const int EH  = E_ * NHEAD;
    const int nTiles = N_ / 16;           // 3125
    const int tilesPerBlock = 25;
    const int gemmGrid  = (nTiles + tilesPerBlock - 1) / tilesPerBlock;
    const int initGrid  = (NH + 255) / 256;
    const int edgeGrid  = (EH + 255) / 256;
    const int statGrid  = (N_ + 1023) / 1024;
    const int elemGrid  = (N_ * 64 + 255) / 256;

    const float* Xin = x;
    float* Xout = Xa;
    for (int l = 0; l < 3; ++l) {
        const int base = 8 + 11 * l;  // sorted keys: We,Wk,Wq,Ws,Wv,bk,bn_b,bn_g,bq,bs,bv
        const float* We_  = (const float*)d_in[base + 0];
        const float* Wk_  = (const float*)d_in[base + 1];
        const float* Wq_  = (const float*)d_in[base + 2];
        const float* Ws_  = (const float*)d_in[base + 3];
        const float* Wv_  = (const float*)d_in[base + 4];
        const float* bk_  = (const float*)d_in[base + 5];
        const float* bnb_ = (const float*)d_in[base + 6];
        const float* bng_ = (const float*)d_in[base + 7];
        const float* bq_  = (const float*)d_in[base + 8];
        const float* bs_  = (const float*)d_in[base + 9];
        const float* bv_  = (const float*)d_in[base + 10];

        init_layer_kernel<<<initGrid, 256, 0, stream>>>(amax, den, sums, NH);
        gemm_qkvs_kernel<<<gemmGrid, 256, 0, stream>>>(Xin, Wq_, Wk_, Wv_, Ws_,
                                                       bq_, bk_, bv_, bs_,
                                                       qkvs, nTiles, tilesPerBlock);
        edge_logits_kernel<<<edgeGrid, 256, 0, stream>>>(ei, ew, qkvs, We_, alpha, amax, E_);
        edge_exp_kernel<<<edgeGrid, 256, 0, stream>>>(ei, alpha, amax, den, E_);
        edge_scatter_kernel<<<edgeGrid, 256, 0, stream>>>(ei, ew, qkvs, We_, alpha, den, E_);
        bn_stats_kernel<<<statGrid, 256, 0, stream>>>(qkvs, sums, N_);
        bn_apply_kernel<<<elemGrid, 256, 0, stream>>>(qkvs, sums, bng_, bnb_, Xout, N_);

        Xin = Xout;
        Xout = (Xout == Xa) ? Xb : Xa;
    }

    init_pool_kernel<<<(GRAPHS * 64 + 255) / 256, 256, 0, stream>>>(pooled, cnt);
    pool_kernel<<<elemGrid, 256, 0, stream>>>(Xin, batch, pooled, cnt, N_);
    head_kernel<<<1, 256, 0, stream>>>(pooled, cnt, fcW1, fcb1, fcW2, fcb2, (float*)d_out);
    (void)out_size;
}
